// CausalSelfAttention_29489245454824
// MI455X (gfx1250) — matrix operand order repaired and checked by reading, not compiled, not run on hardware
//
#include <hip/hip_runtime.h>

typedef __attribute__((ext_vector_type(16))) __bf16 v16bf;
typedef __attribute__((ext_vector_type(8)))  __bf16 v8bf;
typedef __attribute__((ext_vector_type(8)))  float  v8f;

#define BB    4
#define TT    2048
#define CC    1024
#define NH    16
#define HSZ   64
#define MROWS (BB * TT)   /* 8192 */
#define NQKV  (3 * CC)    /* 3072 */

static __device__ __forceinline__ v16bf bf_combine(v8bf lo, v8bf hi) {
  v16bf r;
#pragma unroll
  for (int i = 0; i < 8; ++i) { r[i] = lo[i]; r[8 + i] = hi[i]; }
  return r;
}

// A fragment (16x32 bf16). Lanes 0-15 row M=lane, K {k0..k0+7, k0+16..k0+23};
// lanes 16-31 row M=lane-16, K {k0+8..k0+15, k0+24..k0+31}. Source row-major [M][K].
static __device__ __forceinline__ v16bf load_a_frag(const __bf16* base, int ld, int k0) {
  int lane = threadIdx.x & 31;
  int hf = lane >> 4;
  int r  = lane & 15;
  const __bf16* p = base + (long)r * ld + k0 + hf * 8;
  v8bf c0 = *(const v8bf*)p;
  v8bf c1 = *(const v8bf*)(p + 16);
  return bf_combine(c0, c1);
}

// B fragment (32x16 bf16). Lanes 0-15 col N=lane, K {k0..k0+7, k0+16..k0+23};
// lanes 16-31 col N=lane-16, K {k0+8..k0+15, k0+24..k0+31}. Source is [N][K] row-major (pre-transposed).
static __device__ __forceinline__ v16bf load_b_frag(const __bf16* base, int ld, int k0) {
  int lane = threadIdx.x & 31;
  int hf = lane >> 4;
  int r  = lane & 15;
  const __bf16* p = base + (long)r * ld + k0 + hf * 8;
  v8bf c0 = *(const v8bf*)p;
  v8bf c1 = *(const v8bf*)(p + 16);
  return bf_combine(c0, c1);
}

static __device__ __forceinline__ v8f wmma_bf16(v16bf a, v16bf b, v8f c) {
  return __builtin_amdgcn_wmma_f32_16x16x32_bf16(false, a, false, b, (short)0, c, false, false);
}

static __device__ __forceinline__ float hmax16(float v) {
  v = fmaxf(v, __shfl_xor(v, 1));
  v = fmaxf(v, __shfl_xor(v, 2));
  v = fmaxf(v, __shfl_xor(v, 4));
  v = fmaxf(v, __shfl_xor(v, 8));
  return v;
}
static __device__ __forceinline__ float hsum16(float v) {
  v += __shfl_xor(v, 1);
  v += __shfl_xor(v, 2);
  v += __shfl_xor(v, 4);
  v += __shfl_xor(v, 8);
  return v;
}

// ---------------- prep kernels ----------------
__global__ void cvt_f32_bf16(const float* __restrict__ in, __bf16* __restrict__ out, int n) {
  int i = blockIdx.x * blockDim.x + threadIdx.x;
  int stride = gridDim.x * blockDim.x;
  for (; i < n; i += stride) out[i] = (__bf16)in[i];
}

// w[K][N] fp32 -> wt[N][K] bf16
__global__ void transpose_to_bf16(const float* __restrict__ w, __bf16* __restrict__ wt,
                                  int K, int N) {
  int i = blockIdx.x * blockDim.x + threadIdx.x;
  int total = K * N;
  int stride = gridDim.x * blockDim.x;
  for (; i < total; i += stride) {
    int k = i / N;
    int n = i - k * N;
    wt[(long)n * K + k] = (__bf16)w[i];
  }
}

// ---------------- GEMM1: qkv = x @ w_attn, 32x64 tile/wave, scattered to head layouts ----------------
__global__ __launch_bounds__(256) void gemm_qkv(const __bf16* __restrict__ Xb,
                                                const __bf16* __restrict__ WT,
                                                __bf16* __restrict__ Qb,
                                                __bf16* __restrict__ Kb,
                                                __bf16* __restrict__ VTb) {
  const int Kd = CC;
  int warp = threadIdx.x >> 5;
  int lane = threadIdx.x & 31;
  int gid = blockIdx.x * 8 + warp;
  const int ntiles = NQKV / 64;  // 48
  int mt = gid / ntiles;
  int nt = gid % ntiles;
  int mbase = mt * 32, nbase = nt * 64;
  const __bf16* aA = Xb + (long)mbase * Kd;
  const __bf16* aB = Xb + (long)(mbase + 16) * Kd;
  v8f acc[2][4] = {};
  for (int k0 = 0; k0 < Kd; k0 += 32) {
    if (k0 + 64 < Kd) {   // pull next-next K-step toward the WGP (global_prefetch_b8)
      __builtin_prefetch(aA + (long)lane * Kd + k0 + 64, 0, 3);
      __builtin_prefetch(WT + (long)(nbase + lane) * Kd + k0 + 64, 0, 3);
      __builtin_prefetch(WT + (long)(nbase + 32 + lane) * Kd + k0 + 64, 0, 3);
    }
    v16bf a0 = load_a_frag(aA, Kd, k0);
    v16bf a1 = load_a_frag(aB, Kd, k0);
#pragma unroll
    for (int s = 0; s < 4; ++s) {
      v16bf b = load_b_frag(WT + (long)(nbase + s * 16) * Kd, Kd, k0);
      acc[0][s] = wmma_bf16(a0, b, acc[0][s]);
      acc[1][s] = wmma_bf16(a1, b, acc[1][s]);
    }
  }
  int hf = lane >> 4, r = lane & 15;
  int sel = nbase / CC;            // 0=q,1=k,2=v (64-wide tile never crosses)
  int h   = (nbase % CC) / HSZ;    // tile is head-aligned
#pragma unroll
  for (int mh = 0; mh < 2; ++mh) {
#pragma unroll
    for (int s = 0; s < 4; ++s) {
      int d = s * 16 + r;          // col within head
#pragma unroll
      for (int j = 0; j < 8; ++j) {
        int m  = mbase + mh * 16 + hf * 8 + j;
        int bb = m >> 11;          // / TT
        int t  = m & (TT - 1);
        __bf16 val = (__bf16)acc[mh][s][j];
        long hh = (long)bb * NH + h;
        if (sel == 0)      Qb [(hh * TT + t) * HSZ + d] = val;
        else if (sel == 1) Kb [(hh * TT + t) * HSZ + d] = val;
        else               VTb[(hh * HSZ + d) * TT + t] = val;   // V transposed [hs][T]
      }
    }
  }
}

// ---------------- Flash-style causal attention ----------------
__global__ __launch_bounds__(256) void attn_kernel(const __bf16* __restrict__ Qb,
                                                   const __bf16* __restrict__ Kb,
                                                   const __bf16* __restrict__ VTb,
                                                   __bf16* __restrict__ Yb) {
  __shared__ __bf16 pl[8][16 * 40];   // per-wave P staging, stride 40 (16B-aligned chunks)
  const int warp = threadIdx.x >> 5;
  const int lane = threadIdx.x & 31;
  const int hf = lane >> 4;
  const int r  = lane & 15;
  int gid = blockIdx.x * 8 + warp;
  int qt = gid & 127;        // query tile within (b,h)
  int bh = gid >> 7;         // 0..63
  int qbase = qt * 16;
  const __bf16* q  = Qb  + (long)bh * TT * HSZ + (long)qbase * HSZ;
  const __bf16* kp = Kb  + (long)bh * TT * HSZ;
  const __bf16* vt = VTb + (long)bh * HSZ * TT;

  v16bf qa0 = load_a_frag(q, HSZ, 0);
  v16bf qa1 = load_a_frag(q, HSZ, 32);

  v8f o0 = {}, o1 = {}, o2 = {}, o3 = {};
  float mrow[8], lrow[8];
#pragma unroll
  for (int j = 0; j < 8; ++j) { mrow[j] = -3.0e38f; lrow[j] = 0.f; }

  const int kend = qbase + 16;
  for (int kb = 0; kb < kend; kb += 32) {
    if (kb + 32 < kend) {   // prefetch next key/value block
      __builtin_prefetch(kp + (long)(kb + 32 + lane) * HSZ, 0, 3);
      __builtin_prefetch(vt + (long)lane * TT + kb + 32, 0, 3);
      __builtin_prefetch(vt + (long)(32 + lane) * TT + kb + 32, 0, 3);
    }
    // S = Q K^T for 32 keys as two 16-key subtiles, contraction over hs=64
    v16bf b00 = load_b_frag(kp + (long)kb * HSZ, HSZ, 0);
    v16bf b01 = load_b_frag(kp + (long)kb * HSZ, HSZ, 32);
    v16bf b10 = load_b_frag(kp + (long)(kb + 16) * HSZ, HSZ, 0);
    v16bf b11 = load_b_frag(kp + (long)(kb + 16) * HSZ, HSZ, 32);
    v8f s0 = {}, s1 = {};
    s0 = wmma_bf16(qa0, b00, s0);
    s0 = wmma_bf16(qa1, b01, s0);
    s1 = wmma_bf16(qa0, b10, s1);
    s1 = wmma_bf16(qa1, b11, s1);
    const bool needMask = (kb + 31 > qbase);
#pragma unroll
    for (int j = 0; j < 8; ++j) {
      int qi = qbase + hf * 8 + j;
      float a0 = s0[j] * 0.125f;   // 1/sqrt(64)
      float a1 = s1[j] * 0.125f;
      if (needMask) {
        if (kb + r > qi)      a0 = -3.0e38f;
        if (kb + 16 + r > qi) a1 = -3.0e38f;
      }
      float rm = hmax16(fmaxf(a0, a1));       // row lives in one 16-lane half
      float nm = fmaxf(mrow[j], rm);
      float cf = __expf(mrow[j] - nm);
      float p0 = __expf(a0 - nm);
      float p1 = __expf(a1 - nm);
      float ps = hsum16(p0 + p1);
      mrow[j] = nm;
      lrow[j] = lrow[j] * cf + ps;
      o0[j] *= cf; o1[j] *= cf; o2[j] *= cf; o3[j] *= cf;
      int row = hf * 8 + j;                   // C-layout -> row-major LDS
      pl[warp][row * 40 + r]      = (__bf16)p0;
      pl[warp][row * 40 + 16 + r] = (__bf16)p1;
    }
    asm volatile("s_wait_dscnt 0x0" ::: "memory");
    // reload P in A-fragment layout (16x32, K = 32 keys)
    const __bf16* pp = &pl[warp][r * 40 + hf * 8];
    v8bf c0 = *(const v8bf*)pp;
    v8bf c1 = *(const v8bf*)(pp + 16);
    v16bf pa = bf_combine(c0, c1);
    // O += P V : B from transposed V [hs][T], 4 tiles of 16 hs-cols
    o0 = wmma_bf16(pa, load_b_frag(vt + (long)(0 * 16) * TT, TT, kb), o0);
    o1 = wmma_bf16(pa, load_b_frag(vt + (long)(1 * 16) * TT, TT, kb), o1);
    o2 = wmma_bf16(pa, load_b_frag(vt + (long)(2 * 16) * TT, TT, kb), o2);
    o3 = wmma_bf16(pa, load_b_frag(vt + (long)(3 * 16) * TT, TT, kb), o3);
  }

  int b = bh >> 4, h = bh & 15;
  __bf16* ybase = Yb + (long)b * TT * CC + (long)h * HSZ;
#pragma unroll
  for (int j = 0; j < 8; ++j) {
    float inv = 1.0f / lrow[j];
    int t = qbase + hf * 8 + j;
    __bf16* yp = ybase + (long)t * CC;
    yp[0 * 16 + r] = (__bf16)(o0[j] * inv);
    yp[1 * 16 + r] = (__bf16)(o1[j] * inv);
    yp[2 * 16 + r] = (__bf16)(o2[j] * inv);
    yp[3 * 16 + r] = (__bf16)(o3[j] * inv);
  }
}

// ---------------- GEMM2: out = y @ w_proj, 32x64 tile/wave, f32 result ----------------
__global__ __launch_bounds__(256) void gemm_proj(const __bf16* __restrict__ Yb,
                                                 const __bf16* __restrict__ WT,
                                                 float* __restrict__ Out) {
  const int Kd = CC, N = CC;
  int warp = threadIdx.x >> 5;
  int lane = threadIdx.x & 31;
  int gid = blockIdx.x * 8 + warp;
  const int ntiles = N / 64;   // 16
  int mt = gid / ntiles;
  int nt = gid % ntiles;
  int mbase = mt * 32, nbase = nt * 64;
  const __bf16* aA = Yb + (long)mbase * Kd;
  const __bf16* aB = Yb + (long)(mbase + 16) * Kd;
  v8f acc[2][4] = {};
  for (int k0 = 0; k0 < Kd; k0 += 32) {
    if (k0 + 64 < Kd) {
      __builtin_prefetch(aA + (long)lane * Kd + k0 + 64, 0, 3);
      __builtin_prefetch(WT + (long)(nbase + lane) * Kd + k0 + 64, 0, 3);
      __builtin_prefetch(WT + (long)(nbase + 32 + lane) * Kd + k0 + 64, 0, 3);
    }
    v16bf a0 = load_a_frag(aA, Kd, k0);
    v16bf a1 = load_a_frag(aB, Kd, k0);
#pragma unroll
    for (int s = 0; s < 4; ++s) {
      v16bf b = load_b_frag(WT + (long)(nbase + s * 16) * Kd, Kd, k0);
      acc[0][s] = wmma_bf16(a0, b, acc[0][s]);
      acc[1][s] = wmma_bf16(a1, b, acc[1][s]);
    }
  }
  int hf = lane >> 4, r = lane & 15;
#pragma unroll
  for (int mh = 0; mh < 2; ++mh) {
#pragma unroll
    for (int j = 0; j < 8; ++j) {
      int m = mbase + mh * 16 + hf * 8 + j;
      float* op = Out + (long)m * N + nbase + r;
      op[0]  = acc[mh][0][j];
      op[16] = acc[mh][1][j];
      op[32] = acc[mh][2][j];
      op[48] = acc[mh][3][j];
    }
  }
}

extern "C" void kernel_launch(void* const* d_in, const int* in_sizes, int n_in,
                              void* d_out, int out_size, void* d_ws, size_t ws_size,
                              hipStream_t stream) {
  const float* x      = (const float*)d_in[0];
  const float* w_attn = (const float*)d_in[1];
  const float* w_proj = (const float*)d_in[2];
  float* out = (float*)d_out;

  char* ws = (char*)d_ws;
  size_t off = 0;
  __bf16* xb  = (__bf16*)(ws + off); off += (size_t)MROWS * CC * 2;  // 16 MB
  __bf16* waT = (__bf16*)(ws + off); off += (size_t)NQKV  * CC * 2;  //  6 MB
  __bf16* wpT = (__bf16*)(ws + off); off += (size_t)CC    * CC * 2;  //  2 MB
  __bf16* qb  = (__bf16*)(ws + off); off += (size_t)MROWS * CC * 2;  // 16 MB
  __bf16* kb  = (__bf16*)(ws + off); off += (size_t)MROWS * CC * 2;  // 16 MB
  __bf16* vtb = (__bf16*)(ws + off); off += (size_t)MROWS * CC * 2;  // 16 MB
  __bf16* yb  = (__bf16*)(ws + off); off += (size_t)MROWS * CC * 2;  // 16 MB

  cvt_f32_bf16<<<2048, 256, 0, stream>>>(x, xb, MROWS * CC);
  transpose_to_bf16<<<2048, 256, 0, stream>>>(w_attn, waT, CC, NQKV);
  transpose_to_bf16<<<1024, 256, 0, stream>>>(w_proj, wpT, CC, CC);
  // (8192/32) M-tiles * 48 N-tiles = 12288 waves / 8 per block
  gemm_qkv<<<1536, 256, 0, stream>>>(xb, waT, qb, kb, vtb);
  // 4*16*128 = 8192 query-tile waves / 8
  attn_kernel<<<1024, 256, 0, stream>>>(qb, kb, vtb, yb);
  // (8192/32) * 16 = 4096 waves / 8
  gemm_proj<<<512, 256, 0, stream>>>(yb, wpT, out);
}
